// SubcenterArcMarginProduct_15659450761250
// MI455X (gfx1250) — matrix operand order repaired and checked by reading, not compiled, not run on hardware
//
#include <hip/hip_runtime.h>
#include <hip/hip_fp16.h>
#include <math.h>

typedef __attribute__((ext_vector_type(16))) _Float16 v16h;
typedef __attribute__((ext_vector_type(8)))  _Float16 v8h;
typedef __attribute__((ext_vector_type(4)))  _Float16 v4h;
typedef __attribute__((ext_vector_type(8)))  float    v8f;

#define NB   512      // batch
#define ND   512      // feature dim
#define NC   100000   // classes
#define NK   3        // subcenters
#define ROWS_PER_BLK 48   // 16 classes * 3 subcenters
#define CLS_PER_BLK  16
#define WSTRIDE 520       // padded half stride for weight tile
#define CSTRIDE 50        // padded float stride for cos tile (256 rows/phase)

__device__ __constant__ float d_S     = 30.0f;
__device__ __constant__ float d_COSM  = 0.8775825618903728f;   // cos(0.5)
__device__ __constant__ float d_SINM  = 0.4794255386042030f;   // sin(0.5)
__device__ __constant__ float d_TH    = -0.8775825618903728f;  // cos(pi-0.5)
__device__ __constant__ float d_MM    = 0.2397127693021015f;   // sin(pi-0.5)*0.5

__device__ __forceinline__ unsigned fkey(float f) {
    unsigned u = __float_as_uint(f);
    return (u & 0x80000000u) ? ~u : (u | 0x80000000u);
}

// ---------------------------------------------------------------------------
// Kernel 1: normalize x rows -> f16, and init per-row accumulators
// ---------------------------------------------------------------------------
__global__ __launch_bounds__(256)
void xnorm_init_kernel(const float* __restrict__ x,
                       _Float16* __restrict__ xnh,
                       float* __restrict__ sumexp,
                       unsigned long long* __restrict__ maxkey,
                       float* __restrict__ logity)
{
    __shared__ float red[256];
    const int b = blockIdx.x;
    const int t = threadIdx.x;
    float a0 = x[b * ND + t];
    float a1 = x[b * ND + t + 256];
    red[t] = a0 * a0 + a1 * a1;
    __syncthreads();
    for (int s = 128; s > 0; s >>= 1) {
        if (t < s) red[t] += red[t + s];
        __syncthreads();
    }
    float inv = 1.0f / fmaxf(sqrtf(red[0]), 1e-12f);
    xnh[b * ND + t]       = (_Float16)(a0 * inv);
    xnh[b * ND + t + 256] = (_Float16)(a1 * inv);
    if (t == 0) {
        sumexp[b] = 0.0f;
        maxkey[b] = 0ull;
        logity[b] = 0.0f;
    }
}

// ---------------------------------------------------------------------------
// Kernel 2: per-block: normalize 48 weight rows -> f16 LDS, WMMA GEMM vs all
// 512 batch rows, max over 3 subcenters, ArcFace margin on label class,
// stream sum-exp + argmax into global accumulators.
// ---------------------------------------------------------------------------
__global__ __launch_bounds__(256)
void arc_main_kernel(const float* __restrict__ w,
                     const _Float16* __restrict__ xnh,
                     const int* __restrict__ label,
                     float* __restrict__ sumexp,
                     unsigned long long* __restrict__ maxkey,
                     float* __restrict__ logity)
{
    __shared__ alignas(16) char smem[51200];  // union: wtile (49920B) / cost (51200B)
    _Float16* wtile = (_Float16*)smem;        // [48][WSTRIDE]
    float*    cost  = (float*)smem;           // [256][CSTRIDE] per phase

    const int tid  = threadIdx.x;
    const int lane = tid & 31;
    const int wv   = tid >> 5;        // wave 0..7
    const int blk  = blockIdx.x;
    const int nloc = lane & 15;
    const int hi   = lane >> 4;       // 0: lanes 0-15, 1: lanes 16-31

    // ---- Step A: load + normalize 6 weight rows per wave, f32 -> f16 LDS ----
    for (int rr = 0; rr < 6; ++rr) {
        int r = wv * 6 + rr;
        const float* src = w + (size_t)(blk * ROWS_PER_BLK + r) * ND;
        float4 f[4];
        float ss = 0.0f;
        #pragma unroll
        for (int c = 0; c < 4; ++c) {
            f[c] = ((const float4*)src)[c * 32 + lane];
            ss += f[c].x * f[c].x + f[c].y * f[c].y + f[c].z * f[c].z + f[c].w * f[c].w;
        }
        #pragma unroll
        for (int m = 16; m > 0; m >>= 1) ss += __shfl_xor(ss, m, 32);
        float inv = 1.0f / fmaxf(sqrtf(ss), 1e-12f);
        #pragma unroll
        for (int c = 0; c < 4; ++c) {
            v4h hv = { (_Float16)(f[c].x * inv), (_Float16)(f[c].y * inv),
                       (_Float16)(f[c].z * inv), (_Float16)(f[c].w * inv) };
            *(v4h*)&wtile[r * WSTRIDE + c * 128 + lane * 4] = hv;
        }
    }
    __syncthreads();

    // ---- Step B: WMMA GEMM. wave owns batch tiles {wv, wv+8, wv+16, wv+24},
    //      3 n-tiles of 16 weight rows each. f16 in, f32 accumulate. ----
    v8f acc[4][3];
    #pragma unroll
    for (int t = 0; t < 4; ++t)
        #pragma unroll
        for (int j = 0; j < 3; ++j)
            #pragma unroll
            for (int v = 0; v < 8; ++v) acc[t][j][v] = 0.0f;

    for (int kc = 0; kc < ND; kc += 32) {
        // B fragments: B is 32(K)x16(N); lane = N (mod 16); K = kc + hi*8 + {0..7, 16..23}
        v16h bf[3];
        #pragma unroll
        for (int j = 0; j < 3; ++j) {
            const _Float16* p = &wtile[(j * 16 + nloc) * WSTRIDE + kc + hi * 8];
            v8h lo = *(const v8h*)p;
            v8h hb = *(const v8h*)(p + 16);
            bf[j] = __builtin_shufflevector(lo, hb, 0,1,2,3,4,5,6,7,8,9,10,11,12,13,14,15);
        }
        #pragma unroll
        for (int t = 0; t < 4; ++t) {
            // A fragment: 16(M)x32(K); lane = M (mod 16); K = kc + hi*8 + {0..7, 16..23}
            int m = (wv + 8 * t) * 16 + nloc;
            const _Float16* ap = xnh + m * ND + kc + hi * 8;
            v8h alo = *(const v8h*)ap;
            v8h ahi = *(const v8h*)(ap + 16);
            v16h af = __builtin_shufflevector(alo, ahi, 0,1,2,3,4,5,6,7,8,9,10,11,12,13,14,15);
            #pragma unroll
            for (int j = 0; j < 3; ++j) {
                acc[t][j] = __builtin_amdgcn_wmma_f32_16x16x32_f16(
                    false, af, false, bf[j], (short)0, acc[t][j], false, false);
            }
        }
    }

    // ---- Steps C+D in two phases of 256 batch rows (LDS reuse of wtile) ----
    #pragma unroll
    for (int phase = 0; phase < 2; ++phase) {
        __syncthreads();   // previous use of smem done (wtile / prior cost reads)
        #pragma unroll
        for (int t2 = 0; t2 < 2; ++t2) {
            int t = phase * 2 + t2;
            #pragma unroll
            for (int j = 0; j < 3; ++j) {
                int lmbase = (wv + 8 * t2) * 16 + 8 * hi;  // local row in [0,256)
                int n = j * 16 + nloc;
                #pragma unroll
                for (int v = 0; v < 8; ++v)
                    cost[(lmbase + v) * CSTRIDE + n] = acc[t][j][v];
            }
        }
        __syncthreads();

        // per-thread: one batch row; max over 3 subcenters, margin, stream out
        int b  = tid + phase * 256;
        int lb = label[b];
        float lsum = 0.0f;
        unsigned long long best = 0ull;
        #pragma unroll
        for (int g = 0; g < CLS_PER_BLK; ++g) {
            float c0 = cost[tid * CSTRIDE + g * 3 + 0];
            float c1 = cost[tid * CSTRIDE + g * 3 + 1];
            float c2 = cost[tid * CSTRIDE + g * 3 + 2];
            float cm = fmaxf(c0, fmaxf(c1, c2));
            int cls = blk * CLS_PER_BLK + g;
            float logit;
            if (cls == lb) {
                float s2 = 1.0f - cm * cm;
                s2 = fminf(fmaxf(s2, 0.0f), 1.0f);
                float sy  = sqrtf(s2);
                float phi = cm * d_COSM - sy * d_SINM;
                phi = (cm - d_TH > 0.0f) ? phi : (cm - d_MM);
                logit = d_S * phi;
                logity[b] = logit;        // unique owner: exactly one (block,thread)
            } else {
                logit = d_S * cm;
            }
            lsum += expf(logit);          // logits <= 30, exp <= 1.07e13: safe in f32
            unsigned long long pk = ((unsigned long long)fkey(logit) << 32) | (unsigned)cls;
            best = pk > best ? pk : best;
        }
        atomicAdd(&sumexp[b], lsum);
        atomicMax(&maxkey[b], best);
    }
}

// ---------------------------------------------------------------------------
// Kernel 3: final loss + accuracy reduction
// ---------------------------------------------------------------------------
__global__ __launch_bounds__(256)
void finalize_kernel(const float* __restrict__ sumexp,
                     const unsigned long long* __restrict__ maxkey,
                     const float* __restrict__ logity,
                     const int* __restrict__ label,
                     float* __restrict__ out)
{
    __shared__ float rl[256];
    __shared__ float rc[256];
    const int t = threadIdx.x;
    float lsum = 0.0f, csum = 0.0f;
    for (int b = t; b < NB; b += 256) {
        lsum += logf(sumexp[b]) - logity[b];      // -(logit_y - logsumexp)
        unsigned pred = (unsigned)(maxkey[b] & 0xffffffffull);
        csum += (pred == (unsigned)label[b]) ? 1.0f : 0.0f;
    }
    rl[t] = lsum; rc[t] = csum;
    __syncthreads();
    for (int s = 128; s > 0; s >>= 1) {
        if (t < s) { rl[t] += rl[t + s]; rc[t] += rc[t + s]; }
        __syncthreads();
    }
    if (t == 0) {
        out[0] = rl[0] / (float)NB;
        out[1] = 100.0f * rc[0] / (float)NB;
    }
}

// ---------------------------------------------------------------------------
extern "C" void kernel_launch(void* const* d_in, const int* in_sizes, int n_in,
                              void* d_out, int out_size, void* d_ws, size_t ws_size,
                              hipStream_t stream)
{
    const float* x      = (const float*)d_in[0];   // [512,512]
    const int*   label  = (const int*)d_in[1];     // [512]
    const float* weight = (const float*)d_in[2];   // [300000,512]
    float* out = (float*)d_out;                    // [loss, prec1]

    char* ws = (char*)d_ws;
    _Float16*           xnh    = (_Float16*)(ws);                         // 512*512*2 = 524288
    float*              sumexp = (float*)(ws + 524288);                   // 2048
    unsigned long long* maxkey = (unsigned long long*)(ws + 524288 + 2048);   // 4096
    float*              logity = (float*)(ws + 524288 + 2048 + 4096);     // 2048

    xnorm_init_kernel<<<NB, 256, 0, stream>>>(x, xnh, sumexp, maxkey, logity);
    arc_main_kernel<<<(NC * NK) / ROWS_PER_BLK, 256, 0, stream>>>(
        weight, xnh, label, sumexp, maxkey, logity);
    finalize_kernel<<<1, 256, 0, stream>>>(sumexp, maxkey, logity, label, out);
}
